// ChunkMHSAModule_11596411699733
// MI455X (gfx1250) — compile-verified, matrix operand order checked by reading, not run on hardware
//
#include <hip/hip_runtime.h>
#include <hip/hip_bf16.h>

typedef __attribute__((ext_vector_type(16))) _Float16 v16h;
typedef __attribute__((ext_vector_type(8)))  _Float16 h8;
typedef __attribute__((ext_vector_type(8)))  float    v8f;
typedef __attribute__((ext_vector_type(4)))  int      v4i;

#define LN_EPS 1e-3f
// B=2, T=2048, D=512, H=8, DH=64  ->  M = B*T = 4096, all GEMMs are [4096x512]x[512x512]

// ---------------------------------------------------------------------------
// Async global->LDS copy (CDNA5 GLOBAL_LOAD_ASYNC_TO_LDS_B128, ASYNCcnt) with
// a synchronous fallback if the toolchain lacks the builtin.
// Probe result: param0 = v4i in AS(1) (global), param1 = LDS side.
// ---------------------------------------------------------------------------
#if __has_builtin(__builtin_amdgcn_global_load_async_to_lds_b128)
#define HAVE_ASYNC_LDS 1
#else
#define HAVE_ASYNC_LDS 0
#endif

__device__ __forceinline__ void cp16_to_lds(void* lds_dst, const void* gsrc) {
#if HAVE_ASYNC_LDS
    __builtin_amdgcn_global_load_async_to_lds_b128(
        (__attribute__((address_space(1))) v4i*)(gsrc),
        (__attribute__((address_space(3))) v4i*)(lds_dst), 0, 0);
#else
    *(h8*)lds_dst = *(const h8*)gsrc;
#endif
}

__device__ __forceinline__ void wait_async_lds() {
#if HAVE_ASYNC_LDS
#if __has_builtin(__builtin_amdgcn_s_wait_asynccnt)
    __builtin_amdgcn_s_wait_asynccnt(0);
#else
    asm volatile("s_wait_asynccnt 0" ::: "memory");
#endif
#endif
}

// ---------------------------------------------------------------------------
// Kernel 1: convert the four 512x512 fp32 weight matrices to f16, transposed
// to N-major so they load directly as WMMA B-fragments (Bt[n][k]).
// Wq/Wk/Wv flat = [d][h*64+dh] = [k][n]; Wo flat = [h*64+dh][d] = [k][n].
// ---------------------------------------------------------------------------
__global__ void prep_weights(const float* __restrict__ Wq, const float* __restrict__ Wk,
                             const float* __restrict__ Wv, const float* __restrict__ Wo,
                             _Float16* __restrict__ Wt) {
    int idx = blockIdx.x * blockDim.x + threadIdx.x;   // 4 * 512*512 threads
    int mat = idx >> 18;                               // 512*512 = 1<<18
    int r   = idx & 262143;
    int k   = r >> 9;
    int n   = r & 511;
    const float* W = (mat == 0) ? Wq : (mat == 1) ? Wk : (mat == 2) ? Wv : Wo;
    Wt[(size_t)mat * 262144 + (size_t)n * 512 + k] = (_Float16)W[(size_t)k * 512 + n];
}

// ---------------------------------------------------------------------------
// Kernel 2: LayerNorm, one wave32 per row of 512, f32 math, f16 output.
// ---------------------------------------------------------------------------
__global__ void layernorm_f16(const float* __restrict__ x, const float* __restrict__ gamma,
                              const float* __restrict__ beta, _Float16* __restrict__ h) {
    const int wave = threadIdx.x >> 5;
    const int lane = threadIdx.x & 31;
    const int row  = blockIdx.x * 8 + wave;            // 4096 rows, grid 512
    const float* xr = x + (size_t)row * 512;
    float v[16];
    float s = 0.f, s2 = 0.f;
    #pragma unroll
    for (int i = 0; i < 16; ++i) {
        float t = xr[lane + 32 * i];
        v[i] = t; s += t; s2 += t * t;
    }
    #pragma unroll
    for (int off = 16; off; off >>= 1) {
        s  += __shfl_xor(s,  off, 32);
        s2 += __shfl_xor(s2, off, 32);
    }
    const float mu   = s  * (1.f / 512.f);
    const float var  = s2 * (1.f / 512.f) - mu * mu;
    const float rstd = rsqrtf(var + LN_EPS);
    _Float16* hr = h + (size_t)row * 512;
    #pragma unroll
    for (int i = 0; i < 16; ++i) {
        int d = lane + 32 * i;
        hr[d] = (_Float16)((v[i] - mu) * rstd * gamma[d] + beta[d]);
    }
}

// ---------------------------------------------------------------------------
// Kernel 3/5: WMMA GEMM  C[M x 512] = A[M x 512](f16) * B[512 x 512] (+bias,+resid)
// B supplied N-major (Bt[n][k]).  Block tile 64(M) x 128(N), 8 waves, each wave
// owns a 32x32 tile = 2x2 v_wmma_f32_16x16x32_f16.  K stepped by 32 through a
// double-buffered LDS stage filled by async global->LDS DMA; one barrier/iter.
// ---------------------------------------------------------------------------
template<bool HAS_RESID>
__global__ __launch_bounds__(256)
void gemm_wmma(const _Float16* __restrict__ A, const _Float16* __restrict__ Bt,
               const float* __restrict__ bias, const float* __restrict__ resid,
               float* __restrict__ C) {
    __shared__ _Float16 As[2][64][40];    // 40-half rows: 16B aligned, bank-staggered
    __shared__ _Float16 Bs[2][128][40];

    const int tid  = threadIdx.x;
    const int lane = tid & 31;
    const int wave = tid >> 5;
    const int wm   = wave & 1;            // 2 wave-rows
    const int wn   = wave >> 1;           // 4 wave-cols
    const int m0   = blockIdx.x * 64;
    const int n0   = blockIdx.y * 128;

    v8f acc[2][2] = {};

    const int lrow = tid >> 2;            // 0..63
    const int lk   = (tid & 3) * 8;       // 0,8,16,24

    auto load_tiles = [&](int buf, int k0) {
        cp16_to_lds(&As[buf][lrow][lk],      &A [(size_t)(m0 + lrow)      * 512 + k0 + lk]);
        cp16_to_lds(&Bs[buf][lrow][lk],      &Bt[(size_t)(n0 + lrow)      * 512 + k0 + lk]);
        cp16_to_lds(&Bs[buf][lrow + 64][lk], &Bt[(size_t)(n0 + 64 + lrow) * 512 + k0 + lk]);
    };

    load_tiles(0, 0);
    wait_async_lds();
    __syncthreads();

    int buf = 0;
    for (int k0 = 0; k0 < 512; k0 += 32) {
        if (k0 + 32 < 512) load_tiles(buf ^ 1, k0 + 32);   // DMA next stage behind WMMA

        // ISA 16-bit A/B fragment layout: lane<16 -> K {0..7,16..23}; lane>=16 -> K {8..15,24..31}
        const int fr = lane & 15;
        const int kq = (lane >> 4) * 8;
        v16h af[2], bf[2];
        #pragma unroll
        for (int mt = 0; mt < 2; ++mt) {
            const _Float16* p = &As[buf][wm * 32 + mt * 16 + fr][0];
            ((h8*)&af[mt])[0] = *(const h8*)(p + kq);
            ((h8*)&af[mt])[1] = *(const h8*)(p + 16 + kq);
        }
        #pragma unroll
        for (int nt = 0; nt < 2; ++nt) {
            const _Float16* p = &Bs[buf][wn * 32 + nt * 16 + fr][0];
            ((h8*)&bf[nt])[0] = *(const h8*)(p + kq);
            ((h8*)&bf[nt])[1] = *(const h8*)(p + 16 + kq);
        }
        #pragma unroll
        for (int mt = 0; mt < 2; ++mt)
            #pragma unroll
            for (int nt = 0; nt < 2; ++nt)
                acc[mt][nt] = __builtin_amdgcn_wmma_f32_16x16x32_f16(
                    false, af[mt], false, bf[nt], (short)0, acc[mt][nt], false, false);

        if (k0 + 32 < 512) {
            wait_async_lds();     // next stage landed in LDS
            __syncthreads();      // all waves done reading current stage + see new stage
        }
        buf ^= 1;
    }

    // C fragment: col = lane&15, VGPR r -> row r + 8*(lane>>4)
    const int col   = lane & 15;
    const int rbase = (lane >> 4) * 8;
    #pragma unroll
    for (int mt = 0; mt < 2; ++mt) {
        #pragma unroll
        for (int nt = 0; nt < 2; ++nt) {
            const int n = n0 + wn * 32 + nt * 16 + col;
            const float b = bias[n];
            #pragma unroll
            for (int r = 0; r < 8; ++r) {
                const int m = m0 + wm * 32 + mt * 16 + rbase + r;
                float val = acc[mt][nt][r] + b;
                if (HAS_RESID) val += resid[(size_t)m * 512 + n];
                C[(size_t)m * 512 + n] = val;
            }
        }
    }
}

// ---------------------------------------------------------------------------
// Kernel 4: band attention, window j in [t-6, t+3] clipped to [0,T).
// One thread per (b,t,h): 10-key softmax in registers, ctx out as f16.
// ---------------------------------------------------------------------------
__global__ void attention_win(const float* __restrict__ q, const float* __restrict__ k,
                              const float* __restrict__ v, _Float16* __restrict__ ctx) {
    const int idx = blockIdx.x * blockDim.x + threadIdx.x;   // B*T*H = 32768
    const int h = idx & 7;
    const int t = (idx >> 3) & 2047;
    const int b = idx >> 14;
    const size_t rowq = ((size_t)(b * 2048 + t)) * 512 + h * 64;

    float sc[10];
    float mx = -3.0e38f;
    #pragma unroll
    for (int jj = 0; jj < 10; ++jj) {
        const int j = t - 6 + jj;
        float s = -3.0e38f;
        if (j >= 0 && j < 2048) {
            const size_t rowk = ((size_t)(b * 2048 + j)) * 512 + h * 64;
            float acc = 0.f;
            #pragma unroll 8
            for (int d = 0; d < 64; ++d) acc += q[rowq + d] * k[rowk + d];
            s = acc * 0.125f;                 // 1/sqrt(64)
        }
        sc[jj] = s;
        mx = fmaxf(mx, s);
    }
    float den = 0.f;
    #pragma unroll
    for (int jj = 0; jj < 10; ++jj) {
        const float p = (sc[jj] > -1.0e38f) ? __expf(sc[jj] - mx) : 0.f;
        sc[jj] = p; den += p;
    }
    const float rden = 1.f / den;
    #pragma unroll 4
    for (int d = 0; d < 64; ++d) {
        float acc = 0.f;
        #pragma unroll
        for (int jj = 0; jj < 10; ++jj) {
            const int j = t - 6 + jj;
            if (j >= 0 && j < 2048)
                acc += sc[jj] * v[((size_t)(b * 2048 + j)) * 512 + h * 64 + d];
        }
        ctx[rowq + d] = (_Float16)(acc * rden);
    }
}

// ---------------------------------------------------------------------------
extern "C" void kernel_launch(void* const* d_in, const int* in_sizes, int n_in,
                              void* d_out, int out_size, void* d_ws, size_t ws_size,
                              hipStream_t stream) {
    (void)in_sizes; (void)n_in; (void)out_size; (void)ws_size;
    const float* x     = (const float*)d_in[0];
    const float* gamma = (const float*)d_in[1];
    const float* beta  = (const float*)d_in[2];
    const float* Wq    = (const float*)d_in[3];
    const float* bq    = (const float*)d_in[4];
    const float* Wk    = (const float*)d_in[5];
    const float* bk    = (const float*)d_in[6];
    const float* Wv    = (const float*)d_in[7];
    const float* bv    = (const float*)d_in[8];
    const float* Wo    = (const float*)d_in[9];
    const float* bo    = (const float*)d_in[10];
    float* out = (float*)d_out;

    // Workspace carve-up (all L2-resident: ~34 MB total)
    char* ws = (char*)d_ws;
    _Float16* Wt  = (_Float16*)ws;  ws += (size_t)4 * 262144 * sizeof(_Float16); // 2 MB
    _Float16* h16 = (_Float16*)ws;  ws += (size_t)4096 * 512 * sizeof(_Float16); // 4 MB
    float* qf = (float*)ws;         ws += (size_t)4096 * 512 * sizeof(float);    // 8 MB
    float* kf = (float*)ws;         ws += (size_t)4096 * 512 * sizeof(float);    // 8 MB
    float* vf = (float*)ws;         ws += (size_t)4096 * 512 * sizeof(float);    // 8 MB
    _Float16* ctx16 = (_Float16*)ws;                                             // 4 MB

    prep_weights <<<4096, 256, 0, stream>>>(Wq, Wk, Wv, Wo, Wt);
    layernorm_f16<<<512,  256, 0, stream>>>(x, gamma, beta, h16);

    dim3 gg(64, 4);  // M/64 x N/128
    gemm_wmma<false><<<gg, 256, 0, stream>>>(h16, Wt + 0 * 262144, bq, nullptr, qf);
    gemm_wmma<false><<<gg, 256, 0, stream>>>(h16, Wt + 1 * 262144, bk, nullptr, kf);
    gemm_wmma<false><<<gg, 256, 0, stream>>>(h16, Wt + 2 * 262144, bv, nullptr, vf);

    attention_win<<<128, 256, 0, stream>>>(qf, kf, vf, ctx16);

    gemm_wmma<true><<<gg, 256, 0, stream>>>(ctx16, Wt + 3 * 262144, bo, x, out);
}
